// MultiHeadAttentionLayer_12171937317446
// MI455X (gfx1250) — matrix operand order repaired and checked by reading, not compiled, not run on hardware
//
#include <hip/hip_runtime.h>
#include <hip/hip_bf16.h>

typedef __attribute__((ext_vector_type(16))) __bf16 v16bf;
typedef __attribute__((ext_vector_type(8)))  float  v8f;

#define D_MODEL 512
#define NHEAD   8
#define DKDIM   64
#define BATCH   4
#define SEQ     2048
#define NTOK    (BATCH * SEQ)   // 8192

__device__ __forceinline__ v8f wmma_bf16(v16bf a, v16bf b, v8f c) {
  // D = A(16x32 bf16) * B(32x16 bf16) + C(16x16 f32)
  return __builtin_amdgcn_wmma_f32_16x16x32_bf16(
      /*neg_a=*/false, a, /*neg_b=*/false, b,
      /*c_mod=*/(short)0, c, /*reuse_a=*/false, /*reuse_b=*/false);
}

// Load a 16-bit A-fragment (16x32): lane holds row (lane&15); for half = lane>>4,
// elements 0-15 cover K = 16*half..16*half+15 (contiguous).
__device__ __forceinline__ v16bf load_a_frag(const __bf16* rowPtr, int kbase, int halfId) {
  union { uint4 q[2]; v16bf v; } u;
  u.q[0] = *(const uint4*)(rowPtr + kbase + 16 * halfId);
  u.q[1] = *(const uint4*)(rowPtr + kbase + 16 * halfId + 8);
  return u.v;
}

// ---------------------------------------------------------------------------
// Stage 1: fp32 -> bf16 conversion
// ---------------------------------------------------------------------------
__global__ void cvt_bf16_kernel(const float* __restrict__ src,
                                __bf16* __restrict__ dst, int n) {
  int i = blockIdx.x * blockDim.x + threadIdx.x;
  if (i < n) dst[i] = (__bf16)src[i];
}

// ---------------------------------------------------------------------------
// Stage 2: fused Q/K/V projection.  out[n][j] = sum_d X[n][d]*W[j][d] + b[j]
// grid.z selects projection.  Each wave computes a 64x32 output block
// (4 M-tiles x 2 N-tiles = 8 accumulators; 1.5 b128 loads per WMMA).
//  mode 0: Q -> Qh[b][h][s][dk]  (scaled by 1/8 = 1/sqrt(64))
//  mode 1: K -> Kh[b][h][s][dk]
//  mode 2: V -> Vt[b][h][dk][s]  (transposed for contiguous B-fragments in PV)
// ---------------------------------------------------------------------------
__global__ void proj_qkv_kernel(
    const __bf16* __restrict__ Xq, const __bf16* __restrict__ Xk,
    const __bf16* __restrict__ Xv,
    const __bf16* __restrict__ Wq, const __bf16* __restrict__ Wk,
    const __bf16* __restrict__ Wv,
    const float* __restrict__ bq, const float* __restrict__ bk,
    const float* __restrict__ bv,
    __bf16* __restrict__ Qh, __bf16* __restrict__ Kh, __bf16* __restrict__ Vt) {
  const int mode = blockIdx.z;
  const __bf16* X   = (mode == 0) ? Xq : (mode == 1) ? Xk : Xv;
  const __bf16* W   = (mode == 0) ? Wq : (mode == 1) ? Wk : Wv;
  const float* bias = (mode == 0) ? bq : (mode == 1) ? bk : bv;

  const int lane   = threadIdx.x & 31;
  const int wid    = threadIdx.x >> 5;
  const int halfId = lane >> 4;
  const int l15    = lane & 15;

  const int m0 = (blockIdx.y * 4 + wid) * 64;  // token tile base (64 rows/wave)
  const int n0 = blockIdx.x * 32;              // output-feature tile base

  const __bf16* xrow[4];
#pragma unroll
  for (int mt = 0; mt < 4; ++mt)
    xrow[mt] = X + (size_t)(m0 + 16 * mt + l15) * D_MODEL;
  const __bf16* wrow0 = W + (size_t)(n0 + l15) * D_MODEL;      // B-frag col j
  const __bf16* wrow1 = W + (size_t)(n0 + 16 + l15) * D_MODEL;

  v8f acc[8] = {};  // [mt*2 + nt]

#pragma unroll 2
  for (int kk = 0; kk < D_MODEL; kk += 32) {
    // B layout: lane holds column; element i <-> K = 16*half + i  => contiguous.
    v16bf b0 = *(const v16bf*)(wrow0 + kk + 16 * halfId);
    v16bf b1 = *(const v16bf*)(wrow1 + kk + 16 * halfId);
#pragma unroll
    for (int mt = 0; mt < 4; ++mt) {
      v16bf a = load_a_frag(xrow[mt], kk, halfId);
      acc[mt * 2 + 0] = wmma_bf16(a, b0, acc[mt * 2 + 0]);
      acc[mt * 2 + 1] = wmma_bf16(a, b1, acc[mt * 2 + 1]);
    }
  }

#pragma unroll
  for (int mt = 0; mt < 4; ++mt) {
#pragma unroll
    for (int nt = 0; nt < 2; ++nt) {
      const int col  = n0 + 16 * nt + l15;        // D layout: N = lane&15
      const float bc = bias[col];
      const int h  = col >> 6;
      const int dk = col & 63;
      v8f a = acc[mt * 2 + nt];
#pragma unroll
      for (int v = 0; v < 8; ++v) {
        const int row = m0 + 16 * mt + v + 8 * halfId;  // D layout: M = v+8*half
        const int bb = row >> 11;                        // /SEQ
        const int ss = row & (SEQ - 1);
        const float val = a[v] + bc;
        if (mode == 0)
          Qh[(((size_t)bb * NHEAD + h) * SEQ + ss) * DKDIM + dk] =
              (__bf16)(val * 0.125f);  // fold 1/sqrt(d_k) into Q
        else if (mode == 1)
          Kh[(((size_t)bb * NHEAD + h) * SEQ + ss) * DKDIM + dk] = (__bf16)val;
        else
          Vt[(((size_t)bb * NHEAD + h) * DKDIM + dk) * SEQ + ss] = (__bf16)val;
      }
    }
  }
}

// ---------------------------------------------------------------------------
// Stage 3: attention.  One wave per 16-query tile of one (b,h).
// Online softmax over abs(masked scores); 8 WMMAs per 32-key block.
// ---------------------------------------------------------------------------
__global__ void attn_kernel(const __bf16* __restrict__ Qh,
                            const __bf16* __restrict__ Kh,
                            const __bf16* __restrict__ Vt,
                            const int* __restrict__ mask,
                            __bf16* __restrict__ O) {
  __shared__ __align__(16) __bf16 Plds[4][16][32];  // per-wave P tile (1KB each)

  const int lane   = threadIdx.x & 31;
  const int wid    = threadIdx.x >> 5;
  const int halfId = lane >> 4;
  const int l15    = lane & 15;

  const int tile = blockIdx.x * 4 + wid;      // 4096 tiles total
  const int qt   = tile & (SEQ / 16 - 1);     // 128 query tiles
  const int h    = (tile >> 7) & (NHEAD - 1);
  const int b    = tile >> 10;

  const size_t bh = (size_t)b * NHEAD + h;
  const __bf16* Qbase = Qh + (bh * SEQ + (size_t)qt * 16) * DKDIM;
  const __bf16* Kbase = Kh + bh * SEQ * DKDIM;
  const __bf16* Vbase = Vt + bh * DKDIM * SEQ;
  const int* mrow = mask + b * SEQ;

  // Q A-fragments (d 0..31 and 32..63), loaded once, reused for all key blocks.
  v16bf qf[2];
  {
    const __bf16* qrow = Qbase + (size_t)l15 * DKDIM;
    qf[0] = load_a_frag(qrow, 0, halfId);
    qf[1] = load_a_frag(qrow, 32, halfId);
  }

  v8f oacc[4] = {};           // 16 queries x 64 features, [f] => feat 16f..16f+15
  float mprev[8], lsum[8];
#pragma unroll
  for (int v = 0; v < 8; ++v) { mprev[v] = -INFINITY; lsum[v] = 0.0f; }

  for (int kb = 0; kb < SEQ; kb += 32) {
    // prefetch next key block of K (per-lane key row) and V (two feature rows)
    if (kb + 32 < SEQ) {
      __builtin_prefetch(Kbase + (size_t)(kb + 32 + l15) * DKDIM, 0, 1);
      __builtin_prefetch(Vbase + (size_t)lane * SEQ + kb + 32, 0, 1);
      __builtin_prefetch(Vbase + (size_t)(32 + lane) * SEQ + kb + 32, 0, 1);
    }

    // ---- scores: two 16x16 tiles (keys kb..kb+15, kb+16..kb+31)
    v8f s0 = {}, s1 = {};
    {
      const __bf16* krow0 = Kbase + (size_t)(kb + l15) * DKDIM;
      const __bf16* krow1 = Kbase + (size_t)(kb + 16 + l15) * DKDIM;
      v16bf kf;
      kf = *(const v16bf*)(krow0 + 16 * halfId);       s0 = wmma_bf16(qf[0], kf, s0);
      kf = *(const v16bf*)(krow0 + 32 + 16 * halfId);  s0 = wmma_bf16(qf[1], kf, s0);
      kf = *(const v16bf*)(krow1 + 16 * halfId);       s1 = wmma_bf16(qf[0], kf, s1);
      kf = *(const v16bf*)(krow1 + 32 + 16 * halfId);  s1 = wmma_bf16(qf[1], kf, s1);
    }
    // mask depends only on key column (N = l15 in D layout)
    const bool mk0 = mrow[kb + l15] != 0;
    const bool mk1 = mrow[kb + 16 + l15] != 0;

    // ---- online softmax on abs(mask ? s : -1e9)  (abs(-1e9) = 1e9)
#pragma unroll
    for (int v = 0; v < 8; ++v) {
      float a0 = mk0 ? fabsf(s0[v]) : 1.0e9f;
      float a1 = mk1 ? fabsf(s1[v]) : 1.0e9f;
      float tm = fmaxf(a0, a1);
#pragma unroll
      for (int off = 8; off >= 1; off >>= 1)
        tm = fmaxf(tm, __shfl_xor(tm, off, 32));   // reduce within 16-lane half
      const float mnew = fmaxf(mprev[v], tm);
      const float c  = __expf(mprev[v] - mnew);
      const float p0 = __expf(a0 - mnew);
      const float p1 = __expf(a1 - mnew);
      float rs = p0 + p1;
#pragma unroll
      for (int off = 8; off >= 1; off >>= 1)
        rs += __shfl_xor(rs, off, 32);
      lsum[v] = lsum[v] * c + rs;
      mprev[v] = mnew;
#pragma unroll
      for (int f = 0; f < 4; ++f) oacc[f][v] *= c;  // rescale accumulators

      const int row = v + 8 * halfId;               // D layout M
      Plds[wid][row][l15]      = (__bf16)p0;
      Plds[wid][row][16 + l15] = (__bf16)p1;
    }
    __syncthreads();

    // ---- reload P as an A-fragment (16 queries x 32 keys)
    v16bf pf;
    {
      union { uint4 q[2]; v16bf v; } u;
      u.q[0] = *(const uint4*)&Plds[wid][l15][16 * halfId];
      u.q[1] = *(const uint4*)&Plds[wid][l15][16 * halfId + 8];
      pf = u.v;
    }
    // ---- PV: B-fragment column = feature, K = key (contiguous in Vt)
#pragma unroll
    for (int f = 0; f < 4; ++f) {
      const __bf16* vrow = Vbase + (size_t)(16 * f + l15) * SEQ + kb + 16 * halfId;
      v16bf vf = *(const v16bf*)vrow;
      oacc[f] = wmma_bf16(pf, vf, oacc[f]);
    }
    __syncthreads();
  }

  // ---- normalize and store to O[b][s][h*64 + feat] (bf16, head-merged)
#pragma unroll
  for (int f = 0; f < 4; ++f) {
#pragma unroll
    for (int v = 0; v < 8; ++v) {
      const int row = qt * 16 + v + 8 * halfId;
      const float val = oacc[f][v] / lsum[v];
      O[((size_t)b * SEQ + row) * D_MODEL + h * DKDIM + 16 * f + l15] = (__bf16)val;
    }
  }
}

// ---------------------------------------------------------------------------
// Stage 4: output projection  out[n][j] = sum_d O[n][d]*Wo[j][d] + bo[j] (fp32)
// 64x32 block per wave, same register blocking as stage 2.
// ---------------------------------------------------------------------------
__global__ void out_proj_kernel(const __bf16* __restrict__ O,
                                const __bf16* __restrict__ Wo,
                                const float* __restrict__ bo,
                                float* __restrict__ out) {
  const int lane   = threadIdx.x & 31;
  const int wid    = threadIdx.x >> 5;
  const int halfId = lane >> 4;
  const int l15    = lane & 15;

  const int m0 = (blockIdx.y * 4 + wid) * 64;
  const int n0 = blockIdx.x * 32;

  const __bf16* xrow[4];
#pragma unroll
  for (int mt = 0; mt < 4; ++mt)
    xrow[mt] = O + (size_t)(m0 + 16 * mt + l15) * D_MODEL;
  const __bf16* wrow0 = Wo + (size_t)(n0 + l15) * D_MODEL;
  const __bf16* wrow1 = Wo + (size_t)(n0 + 16 + l15) * D_MODEL;

  v8f acc[8] = {};
#pragma unroll 2
  for (int kk = 0; kk < D_MODEL; kk += 32) {
    v16bf b0 = *(const v16bf*)(wrow0 + kk + 16 * halfId);
    v16bf b1 = *(const v16bf*)(wrow1 + kk + 16 * halfId);
#pragma unroll
    for (int mt = 0; mt < 4; ++mt) {
      v16bf a = load_a_frag(xrow[mt], kk, halfId);
      acc[mt * 2 + 0] = wmma_bf16(a, b0, acc[mt * 2 + 0]);
      acc[mt * 2 + 1] = wmma_bf16(a, b1, acc[mt * 2 + 1]);
    }
  }

#pragma unroll
  for (int mt = 0; mt < 4; ++mt) {
#pragma unroll
    for (int nt = 0; nt < 2; ++nt) {
      const int col  = n0 + 16 * nt + l15;
      const float bc = bo[col];
      v8f a = acc[mt * 2 + nt];
#pragma unroll
      for (int v = 0; v < 8; ++v) {
        const int row = m0 + 16 * mt + v + 8 * halfId;
        out[(size_t)row * D_MODEL + col] = a[v] + bc;
      }
    }
  }
}

// ---------------------------------------------------------------------------
extern "C" void kernel_launch(void* const* d_in, const int* in_sizes, int n_in,
                              void* d_out, int out_size, void* d_ws, size_t ws_size,
                              hipStream_t stream) {
  const float* qf   = (const float*)d_in[0];
  const float* kf   = (const float*)d_in[1];
  const float* vf   = (const float*)d_in[2];
  const int*   mask = (const int*)d_in[3];
  const float* Wq   = (const float*)d_in[4];
  const float* bq   = (const float*)d_in[5];
  const float* Wk   = (const float*)d_in[6];
  const float* bk   = (const float*)d_in[7];
  const float* Wv   = (const float*)d_in[8];
  const float* bv   = (const float*)d_in[9];
  const float* Wo   = (const float*)d_in[10];
  const float* bo   = (const float*)d_in[11];
  float* out = (float*)d_out;

  // workspace carve-up (all offsets 256B-aligned)
  char* ws = (char*)d_ws;
  size_t off = 0;
  auto carve = [&](size_t bytes) {
    void* p = ws + off;
    off += (bytes + 255) & ~(size_t)255;
    return p;
  };
  const size_t actElems = (size_t)NTOK * D_MODEL;      // 4,194,304
  const size_t wElems   = (size_t)D_MODEL * D_MODEL;   // 262,144
  __bf16* Xq  = (__bf16*)carve(actElems * 2);
  __bf16* Xk  = (__bf16*)carve(actElems * 2);
  __bf16* Xv  = (__bf16*)carve(actElems * 2);
  __bf16* WqB = (__bf16*)carve(wElems * 2);
  __bf16* WkB = (__bf16*)carve(wElems * 2);
  __bf16* WvB = (__bf16*)carve(wElems * 2);
  __bf16* WoB = (__bf16*)carve(wElems * 2);
  __bf16* Qh  = (__bf16*)carve(actElems * 2);
  __bf16* Kh  = (__bf16*)carve(actElems * 2);
  __bf16* Vt  = (__bf16*)carve(actElems * 2);
  __bf16* Obf = (__bf16*)carve(actElems * 2);
  (void)ws_size; (void)in_sizes; (void)n_in; (void)out_size;

  // Stage 1: convert activations + weights to bf16
  const int CB = 256;
  cvt_bf16_kernel<<<(int)((actElems + CB - 1) / CB), CB, 0, stream>>>(qf, Xq, (int)actElems);
  cvt_bf16_kernel<<<(int)((actElems + CB - 1) / CB), CB, 0, stream>>>(kf, Xk, (int)actElems);
  cvt_bf16_kernel<<<(int)((actElems + CB - 1) / CB), CB, 0, stream>>>(vf, Xv, (int)actElems);
  cvt_bf16_kernel<<<(int)((wElems + CB - 1) / CB), CB, 0, stream>>>(Wq, WqB, (int)wElems);
  cvt_bf16_kernel<<<(int)((wElems + CB - 1) / CB), CB, 0, stream>>>(Wk, WkB, (int)wElems);
  cvt_bf16_kernel<<<(int)((wElems + CB - 1) / CB), CB, 0, stream>>>(Wv, WvB, (int)wElems);
  cvt_bf16_kernel<<<(int)((wElems + CB - 1) / CB), CB, 0, stream>>>(Wo, WoB, (int)wElems);

  // Stage 2: Q/K/V projections (fused over grid.z); 64x32 per wave
  dim3 blk(128);
  dim3 gproj(D_MODEL / 32, NTOK / 256, 3);
  proj_qkv_kernel<<<gproj, blk, 0, stream>>>(Xq, Xk, Xv, WqB, WkB, WvB,
                                             bq, bk, bv, Qh, Kh, Vt);

  // Stage 3: attention (4096 query tiles, 4 waves/block)
  dim3 gattn((BATCH * NHEAD * (SEQ / 16)) / 4);
  attn_kernel<<<gattn, blk, 0, stream>>>(Qh, Kh, Vt, mask, Obf);

  // Stage 4: output projection to fp32; 64x32 per wave
  dim3 gout(D_MODEL / 32, NTOK / 256);
  out_proj_kernel<<<gout, blk, 0, stream>>>(Obf, WoB, bo, out);
}
